// Qwen3NextCrossAttention_592705487400
// MI455X (gfx1250) — compile-verified, matrix operand order checked by reading, not run on hardware
//
#include <hip/hip_runtime.h>
#include <hip/hip_bf16.h>

// ---------------- problem constants ----------------
constexpr int B   = 64;
constexpr int T   = 8;
constexpr int D   = 512;
constexpr int S   = 512;
constexpr int NKV = 4;
constexpr int G   = T / NKV;          // 2
constexpr int CHUNK = 64;             // S-chunk processed per iteration
constexpr int NC  = S / CHUNK;        // 8 chunks
constexpr float EPS = 1e-6f;
constexpr float SCALE = 0.044194173824159216f; // D^-0.5

typedef __attribute__((ext_vector_type(16))) _Float16 v16h;
typedef __attribute__((ext_vector_type(4)))  _Float16 v4h;
typedef __attribute__((ext_vector_type(8)))  float    v8f;
typedef __attribute__((ext_vector_type(4)))  unsigned v4u;
typedef __attribute__((ext_vector_type(8)))  int      v8i;
typedef __attribute__((ext_vector_type(4)))  int      v4i;

// ---- LDS layout of attn_kernel (dynamic LDS region assumed to start @ 0) ----
constexpr unsigned LDS_HIST0 = 0;                          // [CHUNK][D] f16, 64 KB
constexpr unsigned LDS_HIST1 = LDS_HIST0 + CHUNK * D * 2;  // double buffer, 64 KB
constexpr unsigned LDS_KV    = LDS_HIST1 + CHUNK * D * 2;  // k rows / vT, 64 KB
constexpr unsigned LDS_QK    = LDS_KV    + CHUNK * D * 2;  // [G][D] f16
constexpr unsigned LDS_ATTN  = LDS_QK    + G * D * 2;      // [G][S] f16
constexpr unsigned LDS_SCORE = LDS_ATTN  + G * S * 2;      // [G][S] f32
constexpr unsigned LDS_RS    = LDS_SCORE + G * S * 4;      // [S] f32
constexpr unsigned LDS_ROWSQ = LDS_RS    + S * 4;          // [CHUNK] f32
constexpr unsigned LDS_RED   = LDS_ROWSQ + CHUNK * 4;      // [128] f32
constexpr unsigned LDS_TOTAL = LDS_RED   + 128 * 4;        // ~203 KB (<320 KB/WGP)

__device__ __forceinline__ v8f wmma32(v16h a, v16h b, v8f c) {
  return __builtin_amdgcn_wmma_f32_16x16x32_f16(false, a, false, b,
                                                (short)0, c, false, false);
}

// K-column (within a 32-wide K tile) of dword-pair p (VGPR p holds K = kcol, kcol+1)
__device__ __forceinline__ int kcol(int p) {
  return ((p & 4) ? 16 : 0) + 2 * (p & 3);
}

// Load a 16(rows) x 32(K) fragment from row-major f16 memory (LDS or global).
// Rows beyond maxRow are clamped (for the G=2 query/attn rows padded to 16).
__device__ __forceinline__ v16h load_ab_f16(const _Float16* src, int pitch,
                                            int lane, int maxRow) {
  int m = lane & 15; if (m > maxRow) m = maxRow;
  int hi = (lane >> 4) & 1;
  const _Float16* row = src + m * pitch + 8 * hi;
  v16h f;
#pragma unroll
  for (int p = 0; p < 8; ++p) {
    union { unsigned u; _Float16 h[2]; } x;
    x.u = *reinterpret_cast<const unsigned*>(row + kcol(p));
    f[2 * p]     = x.h[0];
    f[2 * p + 1] = x.h[1];
  }
  return f;
}

// Same but source is row-major f32 in global memory (converted on the fly).
__device__ __forceinline__ v16h load_ab_f32(const float* src, int pitch, int lane) {
  int m = lane & 15;
  int hi = (lane >> 4) & 1;
  const float* row = src + (size_t)m * pitch + 8 * hi;
  v16h f;
#pragma unroll
  for (int p = 0; p < 8; ++p) {
    const float* q = row + kcol(p);
    f[2 * p]     = (_Float16)q[0];
    f[2 * p + 1] = (_Float16)q[1];
  }
  return f;
}

__device__ __forceinline__ v8f vzero() {
  v8f z = {0.f, 0.f, 0.f, 0.f, 0.f, 0.f, 0.f, 0.f};
  return z;
}

// ---- Tensor Data Mover: DMA one [CHUNK x D] f16 tile (row-major, packed) ----
// D# per CDNA5 ISA ch.8: group0 = count/lds_addr/global_addr/type,
// group1 = data_size, tensor dims, tile dims, stride. Groups 2/3 zero (2D).
__device__ __forceinline__ void tdm_load_chunk(const _Float16* gsrc,
                                               unsigned ldsOff) {
  unsigned long long ga = (unsigned long long)(uintptr_t)gsrc;
  v4u g0;
  g0[0] = 1u;                                   // count=1, user descriptor
  g0[1] = ldsOff;                               // lds_addr (bytes)
  g0[2] = (unsigned)(ga & 0xFFFFFFFFu);         // global_addr[31:0]
  g0[3] = (unsigned)((ga >> 32) & 0x01FFFFFFu)  // global_addr[56:32]
          | (2u << 30);                         // type=2 ("image")
  v8i g1;
  g1[0] = (int)(1u << 16);                      // data_size=1 (2 bytes)
  g1[1] = (int)((unsigned)D << 16);             // tensor_dim0[15:0] in [31:16]
  g1[2] = (int)((unsigned)CHUNK << 16);         // dim0 hi=0 | tensor_dim1 lo
  g1[3] = (int)((unsigned)D << 16);             // dim1 hi=0 | tile_dim0=512
  g1[4] = (int)CHUNK;                           // tile_dim1=64, tile_dim2=0
  g1[5] = (int)D;                               // tensor_dim0_stride lo32
  g1[6] = 0;                                    // stride0 hi | dim1_stride lo
  g1[7] = 0;
  v4i gz = {0, 0, 0, 0};
#if __clang_major__ >= 23
  v8i gz8 = {0, 0, 0, 0, 0, 0, 0, 0};
  __builtin_amdgcn_tensor_load_to_lds(g0, g1, gz, gz, gz8, 0);
#else
  __builtin_amdgcn_tensor_load_to_lds(g0, g1, gz, gz, 0);
#endif
}

// ---------------- kernel 0a: swizzle weights f32 -> f16 B-fragment layout ------
__global__ __launch_bounds__(256)
void swizzle_w(const float* __restrict__ W, _Float16* __restrict__ out, int E) {
  int idx = blockIdx.x * 256 + threadIdx.x;
  if (idx >= E * 512) return;
  int h    = idx & 15;
  int lane = (idx >> 4) & 31;
  int tile = idx >> 9;
  int dt   = tile & 15;       // D/32 == 16
  int etg  = tile >> 4;
  int e = etg * 16 + (lane & 15);
  int d = dt * 32 + 2 * (h & 8) + 8 * (lane >> 4) + (h & 7);
  out[idx] = (_Float16)W[(size_t)e * 512 + d];
}

// ---------------- kernel 0b: history f32 -> packed f16 (TDM source) -----------
__global__ __launch_bounds__(256)
void cvt_hist(const float* __restrict__ src, _Float16* __restrict__ dst, int n4) {
  int i = blockIdx.x * 256 + threadIdx.x;
  if (i >= n4) return;
  float4 x = reinterpret_cast<const float4*>(src)[i];
  v4h h;
  h[0] = (_Float16)x.x; h[1] = (_Float16)x.y;
  h[2] = (_Float16)x.z; h[3] = (_Float16)x.w;
  *reinterpret_cast<v4h*>(dst + (size_t)i * 4) = h;
}

// ---------------- kernel 1: q = rmsnorm(target @ Wq^T) * (1+qw) * (1+kw) ------
__global__ __launch_bounds__(256)
void qproj_kernel(const float* __restrict__ tgt, const _Float16* __restrict__ wq,
                  const float* __restrict__ qw, const float* __restrict__ kw,
                  _Float16* __restrict__ qkOut) {
  __shared__ float qLds[16][D];
  __shared__ float rowSq[16];

  int tid  = threadIdx.x;
  int lane = tid & 31;
  int w    = tid >> 5;
  int rowBase = blockIdx.x * 16;

  v8f acc[4];
#pragma unroll
  for (int j = 0; j < 4; ++j) acc[j] = vzero();

  for (int kt = 0; kt < 16; ++kt) {
    v16h a = load_ab_f32(tgt + (size_t)rowBase * D + kt * 32, D, lane);
#pragma unroll
    for (int j = 0; j < 4; ++j) {
      int etg = w * 4 + j;
      const _Float16* tb = wq + ((size_t)(etg * 16 + kt)) * 512;
      v16h bf = *reinterpret_cast<const v16h*>(tb + lane * 16);
      acc[j] = wmma32(a, bf, acc[j]);
    }
  }

  int nn = lane & 15, mb = (lane >> 4) * 8;
#pragma unroll
  for (int j = 0; j < 4; ++j) {
    int e = (w * 4 + j) * 16 + nn;
#pragma unroll
    for (int r = 0; r < 8; ++r) qLds[mb + r][e] = acc[j][r];
  }
  if (tid < 16) rowSq[tid] = 0.f;
  __syncthreads();

  {
    int r = tid >> 4, seg = tid & 15;
    float s = 0.f;
    for (int i = 0; i < 32; ++i) {
      float x = qLds[r][seg * 32 + i];
      s += x * x;
    }
    atomicAdd(&rowSq[r], s);
  }
  __syncthreads();

  for (int i = tid; i < 16 * D; i += 256) {
    int r = i >> 9, col = i & 511;
    float rs = rsqrtf(rowSq[r] * (1.f / D) + EPS);
    float v = qLds[r][col] * rs * (1.f + qw[col]) * (1.f + kw[col]);
    qkOut[(size_t)(rowBase + r) * D + col] = (_Float16)v;
  }
}

// ---------------- kernel 2: fused kv-proj + attention, TDM-pipelined ----------
__global__ __launch_bounds__(256)
void attn_kernel(const _Float16* __restrict__ histF16,
                 const int* __restrict__ mask,
                 const _Float16* __restrict__ swk, const _Float16* __restrict__ swv,
                 const _Float16* __restrict__ qkG,
                 float* __restrict__ outTok, float* __restrict__ outAttn) {
  extern __shared__ __align__(16) char smemRaw[];
  _Float16* kvL    = (_Float16*)(smemRaw + LDS_KV);
  _Float16* qkL    = (_Float16*)(smemRaw + LDS_QK);
  _Float16* attnL  = (_Float16*)(smemRaw + LDS_ATTN);
  float*    scoresL = (float*)(smemRaw + LDS_SCORE);
  float*    rsAll   = (float*)(smemRaw + LDS_RS);
  float*    rowSq   = (float*)(smemRaw + LDS_ROWSQ);
  float*    red     = (float*)(smemRaw + LDS_RED);

  int wg = blockIdx.x;
  int b  = wg / NKV;
  int n  = wg % NKV;
  int tid  = threadIdx.x;
  int lane = tid & 31;
  int w    = tid >> 5;
  int nn = lane & 15;
  int mb = (lane >> 4) * 8;

  const _Float16* histSrc = histF16 + (size_t)b * S * D;

  // prologue: kick off DMA of chunk 0 while staging q rows
  if (w == 0) tdm_load_chunk(histSrc, LDS_HIST0);
  for (int i = tid; i < G * D; i += 256)
    qkL[i] = qkG[(size_t)(b * T + n * G) * D + i];

  // ================= pass 1: k projection + scores =================
  for (int c = 0; c < NC; ++c) {
    const _Float16* histL =
        (const _Float16*)(smemRaw + ((c & 1) ? LDS_HIST1 : LDS_HIST0));
    if (w == 0) {
      if (c + 1 < NC) {
        tdm_load_chunk(histSrc + (size_t)(c + 1) * CHUNK * D,
                       ((c + 1) & 1) ? LDS_HIST1 : LDS_HIST0);
        __builtin_amdgcn_s_wait_tensorcnt(1);   // chunk c landed
      } else {
        __builtin_amdgcn_s_wait_tensorcnt(0);
      }
    }
    for (int i = tid; i < CHUNK; i += 256) rowSq[i] = 0.f;
    __syncthreads();

    // k-proj: wave w -> mTile = w/2, nHalf = w&1 (16 N-tiles)
    {
      int mT = w >> 1, nHalf = w & 1;
      v8f acc[16];
#pragma unroll
      for (int j = 0; j < 16; ++j) acc[j] = vzero();
      for (int kt = 0; kt < 16; ++kt) {
        v16h a = load_ab_f16(histL + (mT * 16) * D + kt * 32, D, lane, 15);
#pragma unroll
        for (int j = 0; j < 16; ++j) {
          int etg = n * 32 + nHalf * 16 + j;   // tile row in full E=2048
          const _Float16* tb = swk + ((size_t)(etg * 16 + kt)) * 512;
          v16h bf = *reinterpret_cast<const v16h*>(tb + lane * 16);
          acc[j] = wmma32(a, bf, acc[j]);
        }
      }
      // store raw k rows row-major [s][d]
#pragma unroll
      for (int j = 0; j < 16; ++j) {
        int e = (nHalf * 16 + j) * 16 + nn;
#pragma unroll
        for (int r = 0; r < 8; ++r) {
          int s = mT * 16 + mb + r;
          kvL[s * D + e] = (_Float16)acc[j][r];
        }
      }
    }
    __syncthreads();

    // per-row sum of squares (raw k) -> rs_k
    {
      int srow = tid >> 2, q = tid & 3;
      float ssum = 0.f;
      const _Float16* kr = kvL + srow * D + q * 128;
      for (int i = 0; i < 128; ++i) {
        float x = (float)kr[i];
        ssum += x * x;
      }
      atomicAdd(&rowSq[srow], ssum);
    }
    __syncthreads();
    for (int i = tid; i < CHUNK; i += 256)
      rsAll[c * CHUNK + i] = rsqrtf(rowSq[i] * (1.f / D) + EPS);
    __syncthreads();

    // scores: waves 0..3 each take one 16-wide s-tile
    if (w < 4) {
      v8f acc = vzero();
      for (int kt = 0; kt < 16; ++kt) {
        v16h a = load_ab_f16(qkL + kt * 32, D, lane, G - 1);
        v16h bf = load_ab_f16(kvL + (w * 16) * D + kt * 32, D, lane, 15);
        acc = wmma32(a, bf, acc);
      }
      if (lane < 16) {
        int s = c * CHUNK + w * 16 + lane;
        scoresL[0 * S + s] = acc[0];   // g = 0
        scoresL[1 * S + s] = acc[1];   // g = 1
      }
    }
    __syncthreads();
  }

  // ================= softmax over S =================
  for (int i = tid; i < G * S; i += 256) {
    int s = i & (S - 1);
    float v = scoresL[i] * rsAll[s] * SCALE;
    if (mask[b * S + s]) v = -__builtin_inff();
    scoresL[i] = v;
  }
  __syncthreads();
  if (tid < 64) {
    int g = tid >> 5, l = tid & 31;
    float mx = -__builtin_inff();
    for (int s = l; s < S; s += 32) mx = fmaxf(mx, scoresL[g * S + s]);
    red[g * 32 + l] = mx;
  }
  __syncthreads();
  if (tid < 2) {
    float mx = -__builtin_inff();
    for (int l = 0; l < 32; ++l) mx = fmaxf(mx, red[tid * 32 + l]);
    red[tid * 32] = mx;
  }
  __syncthreads();
  if (tid < 64) {
    int g = tid >> 5, l = tid & 31;
    float mx = red[g * 32];
    float sm = 0.f;
    for (int s = l; s < S; s += 32) {
      float e = __expf(scoresL[g * S + s] - mx);
      scoresL[g * S + s] = e;
      sm += e;
    }
    red[64 + g * 32 + l] = sm;
  }
  __syncthreads();
  if (tid < 2) {
    float sm = 0.f;
    for (int l = 0; l < 32; ++l) sm += red[64 + tid * 32 + l];
    red[64 + tid * 32] = 1.f / sm;
  }
  __syncthreads();
  for (int i = tid; i < G * S; i += 256) {
    int g = i >> 9, s = i & (S - 1);
    float a = scoresL[i] * red[64 + g * 32];
    attnL[i] = (_Float16)a;
    outAttn[(size_t)(b * T + n * G + g) * S + s] = a;
  }
  __syncthreads();

  // ================= pass 2: v projection + out = attn @ v =================
  v8f oacc[4];
#pragma unroll
  for (int j = 0; j < 4; ++j) oacc[j] = vzero();

  if (w == 0) tdm_load_chunk(histSrc, LDS_HIST0);

  for (int c = 0; c < NC; ++c) {
    const _Float16* histL =
        (const _Float16*)(smemRaw + ((c & 1) ? LDS_HIST1 : LDS_HIST0));
    if (w == 0) {
      if (c + 1 < NC) {
        tdm_load_chunk(histSrc + (size_t)(c + 1) * CHUNK * D,
                       ((c + 1) & 1) ? LDS_HIST1 : LDS_HIST0);
        __builtin_amdgcn_s_wait_tensorcnt(1);
      } else {
        __builtin_amdgcn_s_wait_tensorcnt(0);
      }
    }
    __syncthreads();

    // v-proj, stored transposed: vT[e][s] (contiguous 8 halves per lane)
    {
      int mT = w >> 1, nHalf = w & 1;
      v8f acc[16];
#pragma unroll
      for (int j = 0; j < 16; ++j) acc[j] = vzero();
      for (int kt = 0; kt < 16; ++kt) {
        v16h a = load_ab_f16(histL + (mT * 16) * D + kt * 32, D, lane, 15);
#pragma unroll
        for (int j = 0; j < 16; ++j) {
          int etg = n * 32 + nHalf * 16 + j;
          const _Float16* tb = swv + ((size_t)(etg * 16 + kt)) * 512;
          v16h bf = *reinterpret_cast<const v16h*>(tb + lane * 16);
          acc[j] = wmma32(a, bf, acc[j]);
        }
      }
#pragma unroll
      for (int j = 0; j < 16; ++j) {
        int e = (nHalf * 16 + j) * 16 + nn;
        _Float16* dst = kvL + e * CHUNK + mT * 16 + mb;
#pragma unroll
        for (int r = 0; r < 8; ++r) dst[r] = (_Float16)acc[j][r];
      }
    }
    __syncthreads();

    // out += attn[:, chunk] @ v_chunk ; wave w owns e range [w*64, w*64+64)
#pragma unroll
    for (int ks = 0; ks < CHUNK / 32; ++ks) {
      v16h a = load_ab_f16(attnL + c * CHUNK + ks * 32, S, lane, G - 1);
#pragma unroll
      for (int j = 0; j < 4; ++j) {
        const _Float16* bsrc = kvL + (w * 64 + j * 16) * CHUNK + ks * 32;
        v16h bf = load_ab_f16(bsrc, CHUNK, lane, 15);
        oacc[j] = wmma32(a, bf, oacc[j]);
      }
    }
    __syncthreads();
  }

  // write tokens: valid rows g=0 (VGPR0) and g=1 (VGPR1), lanes 0..15
  if (lane < 16) {
#pragma unroll
    for (int j = 0; j < 4; ++j) {
      int e = w * 64 + j * 16 + lane;
      outTok[(size_t)(b * T + n * G + 0) * D + e] = oacc[j][0];
      outTok[(size_t)(b * T + n * G + 1) * D + e] = oacc[j][1];
    }
  }
}

// ---------------- host launcher ----------------
extern "C" void kernel_launch(void* const* d_in, const int* in_sizes, int n_in,
                              void* d_out, int out_size, void* d_ws, size_t ws_size,
                              hipStream_t stream) {
  const float* tgt  = (const float*)d_in[0];
  const float* hist = (const float*)d_in[1];
  const int*   mask = (const int*)d_in[2];
  const float* wq   = (const float*)d_in[3];
  const float* wk   = (const float*)d_in[4];
  const float* wv   = (const float*)d_in[5];
  const float* qnw  = (const float*)d_in[6];
  const float* knw  = (const float*)d_in[7];

  float* outTok  = (float*)d_out;
  float* outAttn = outTok + (size_t)B * T * D;

  char* ws = (char*)d_ws;
  _Float16* swq  = (_Float16*)(ws);
  _Float16* swk  = (_Float16*)(ws + (size_t)512 * 1024);
  _Float16* swv  = (_Float16*)(ws + (size_t)512 * 1024 + 2u * 1024 * 1024);
  _Float16* qk   = (_Float16*)(ws + (size_t)512 * 1024 + 4u * 1024 * 1024);
  _Float16* h16  = (_Float16*)(ws + (size_t)5 * 1024 * 1024);  // 32 MB

  swizzle_w<<<(512 * 512) / 256, 256, 0, stream>>>(wq, swq, 512);
  swizzle_w<<<(2048 * 512) / 256, 256, 0, stream>>>(wk, swk, 2048);
  swizzle_w<<<(2048 * 512) / 256, 256, 0, stream>>>(wv, swv, 2048);

  int n4 = B * S * D / 4;
  cvt_hist<<<(n4 + 255) / 256, 256, 0, stream>>>(hist, h16, n4);

  qproj_kernel<<<B * T / 16, 256, 0, stream>>>(tgt, swq, qnw, knw, qk);

  attn_kernel<<<B * NKV, 256, LDS_TOTAL, stream>>>(h16, mask, swk, swv, qk,
                                                   outTok, outAttn);
}
